// ModifiedMultiHeadAttention_69982197121407
// MI455X (gfx1250) — matrix operand order repaired and checked by reading, not compiled, not run on hardware
//
#include <hip/hip_runtime.h>

// ---------------- problem constants (fixed shapes) ----------------
constexpr int Bc  = 2;
constexpr int Sc  = 2048;
constexpr int Dc  = 1024;
constexpr int Hc  = 16;
constexpr int HDc = 64;     // head dim
constexpr int ADc = 128;    // adapter dim
constexpr int NTc = Bc * Sc;        // 4096 tokens
constexpr size_t NTD = (size_t)NTc * Dc;   // 4,194,304

// ---------------- WMMA fragment types (gfx1250, wave32) ----------------
typedef __attribute__((ext_vector_type(16))) __bf16 v16bf;
typedef __attribute__((ext_vector_type(8)))  float  v8f;

union FragAB { v16bf v; uint4 q[2]; unsigned short h[16]; };
union FragC  { v8f  v; float f[8]; };

__device__ __forceinline__ unsigned short f2bf(float f) {
  union { float f; unsigned u; } x; x.f = f;
  unsigned r = x.u + 0x7FFFu + ((x.u >> 16) & 1u);   // round-to-nearest-even
  return (unsigned short)(r >> 16);
}

// load 16 bf16 as two 16-byte chunks
__device__ __forceinline__ void load_frag16(FragAB& fr,
                                            const unsigned short* p0,
                                            const unsigned short* p1) {
  fr.q[0] = *reinterpret_cast<const uint4*>(p0);
  fr.q[1] = *reinterpret_cast<const uint4*>(p1);
}

// ---------------- fp32 -> bf16 conversion ----------------
__global__ void cvt_bf16_kernel(const float* __restrict__ src,
                                unsigned short* __restrict__ dst, int n) {
  int i = blockIdx.x * blockDim.x + threadIdx.x;
  int stride = gridDim.x * blockDim.x;
  for (; i < n; i += stride) dst[i] = f2bf(src[i]);
}

// ---------------- generic bf16 GEMM: Y = A(MxK) * W(NxK)^T + bias ----------------
// modes for the epilogue
#define MODE_BF16     0   // outU[row*N+col] = bf16(val)
#define MODE_VT       1   // per-head transposed V store: Vt[(b,h,hd)][s]
#define MODE_F32BF16  2   // outF + outU
#define MODE_GELU     3   // exact-erf GELU -> bf16
#define MODE_RESID    4   // outF = baseF + val   (fp32 final output)

__global__ __launch_bounds__(128) void gemm_bf16_kernel(
    const unsigned short* __restrict__ A, int lda,
    const unsigned short* __restrict__ W, int ldw,
    const float* __restrict__ bias,
    int M, int N, int K, int mode,
    unsigned short* __restrict__ outU,
    float* __restrict__ outF,
    const float* __restrict__ baseF)
{
  const int lane = threadIdx.x & 31;
  const int wave = threadIdx.x >> 5;
  const int m_id = lane & 15;      // row (A) / col (B) index within fragment
  const int g    = lane >> 4;      // lane group (K-half selector)
  const int row0 = (blockIdx.y * 4 + wave) * 32;
  const int col0 = blockIdx.x * 64;
  if (row0 >= M) return;

  FragC acc0[4], acc1[4];
#pragma unroll
  for (int c = 0; c < 4; ++c)
#pragma unroll
    for (int j = 0; j < 8; ++j) { acc0[c].f[j] = 0.f; acc1[c].f[j] = 0.f; }

  for (int k0 = 0; k0 < K; k0 += 32) {
    // A fragments: 16x32, lane m holds row, K chunks at k0+g*8 and k0+16+g*8
    const unsigned short* ap = A + (size_t)(row0 + m_id) * lda + k0 + g * 8;
    FragAB a0, a1;
    load_frag16(a0, ap, ap + 16);
    const unsigned short* ap1 = ap + (size_t)16 * lda;
    load_frag16(a1, ap1, ap1 + 16);
    __builtin_prefetch(ap + 32, 0, 1);            // global_prefetch_b8
#pragma unroll
    for (int c = 0; c < 4; ++c) {
      // B fragment: 32x16 (K x N). lane n=m_id holds col, K chunks at k0+g*8 and k0+16+g*8
      const unsigned short* bp =
          W + (size_t)(col0 + c * 16 + m_id) * ldw + k0 + g * 8;
      FragAB bf;
      load_frag16(bf, bp, bp + 16);
      acc0[c].v = __builtin_amdgcn_wmma_f32_16x16x32_bf16(
          false, a0.v, false, bf.v, (short)0, acc0[c].v, false, false);
      acc1[c].v = __builtin_amdgcn_wmma_f32_16x16x32_bf16(
          false, a1.v, false, bf.v, (short)0, acc1[c].v, false, false);
    }
  }

  // epilogue
  auto epi = [&](FragC (&acc)[4], int rowbase) {
#pragma unroll
    for (int c = 0; c < 4; ++c) {
      const int col = col0 + c * 16 + m_id;
      const float bv = bias ? bias[col] : 0.f;
#pragma unroll
      for (int j = 0; j < 8; ++j) {
        const int row = rowbase + j + 8 * g;
        float val = acc[c].f[j] + bv;
        const size_t idx = (size_t)row * N + col;
        if (mode == MODE_BF16) {
          outU[idx] = f2bf(val);
        } else if (mode == MODE_VT) {
          const int bb = row >> 11, ss = row & (Sc - 1);
          const int hh = col >> 6,  hd = col & (HDc - 1);
          outU[((size_t)((bb * Hc + hh) * HDc + hd)) * Sc + ss] = f2bf(val);
        } else if (mode == MODE_F32BF16) {
          outF[idx] = val; outU[idx] = f2bf(val);
        } else if (mode == MODE_GELU) {
          float gv = 0.5f * val * (1.f + erff(val * 0.70710678118654752f));
          outU[idx] = f2bf(gv);
        } else { // MODE_RESID
          outF[idx] = baseF[idx] + val;
        }
      }
    }
  };
  epi(acc0, row0);
  epi(acc1, row0 + 16);
}

// ---------------- flash attention ----------------
// One block = 4 waves, all on the SAME (b,h); each wave owns a 16-query tile.
// Per 32-key step the block cooperatively stages the shared K tile (32x64) and
// V tile (64x32, per-head transposed) into LDS with global_load_async_to_lds,
// then all 4 waves consume them (4x global-traffic reduction).
// Q,K: bf16 [NT, D] (head h occupies cols h*64..h*64+63)
// Vt : bf16 [B*H*HD, S]
// ctx: bf16 [NT, D]
__global__ __launch_bounds__(128) void attn_kernel(
    const unsigned short* __restrict__ Q,
    const unsigned short* __restrict__ Kb,
    const unsigned short* __restrict__ Vt,
    unsigned short* __restrict__ ctx)
{
  __shared__ __align__(16) unsigned short Ksh[32 * 64];   // [key][hd]   4 KB
  __shared__ __align__(16) unsigned short Vsh[64 * 32];   // [hd][key]   4 KB
  __shared__ __align__(16) unsigned short Pst[4][16 * 40]; // per-wave P tile

  const int tid  = threadIdx.x;
  const int lane = tid & 31;
  const int wave = tid >> 5;
  const int bh = blockIdx.x >> 5;                  // uniform per block
  const int qt = (blockIdx.x & 31) * 4 + wave;     // query tile 0..127
  const int b = bh >> 4, h = bh & 15;
  const int m_id = lane & 15, g = lane >> 4;

  // Q fragments: rows = 16 queries, K-dim = head dim (two 32-chunks)
  FragAB qa[2];
  {
    const size_t tokQ = (size_t)(b * Sc + qt * 16 + m_id);
#pragma unroll
    for (int c2 = 0; c2 < 2; ++c2) {
      const unsigned short* qp = Q + tokQ * Dc + h * HDc + c2 * 32 + g * 16;
      load_frag16(qa[c2], qp, qp + 8);
    }
  }

  FragC o[4];
  float mrun[8], lrun[8];
#pragma unroll
  for (int c = 0; c < 4; ++c)
#pragma unroll
    for (int j = 0; j < 8; ++j) o[c].f[j] = 0.f;
#pragma unroll
  for (int j = 0; j < 8; ++j) { mrun[j] = -3.0e38f; lrun[j] = 0.f; }

  const size_t vtbase = (size_t)((b * Hc + h) * HDc) * Sc;
  unsigned short* pl = &Pst[wave][0];

  // per-lane async-copy geometry (two 16B chunks per lane per tile)
  const int krow0 = (tid * 2) >> 3,     kcol0 = ((tid * 2)     & 7) * 8;
  const int krow1 = (tid * 2 + 1) >> 3, kcol1 = ((tid * 2 + 1) & 7) * 8;
  const int vrow0 = (tid * 2) >> 2,     vcol0 = ((tid * 2)     & 3) * 8;
  const int vrow1 = (tid * 2 + 1) >> 2, vcol1 = ((tid * 2 + 1) & 3) * 8;
  const unsigned kld0 = (unsigned)(uintptr_t)&Ksh[krow0 * 64 + kcol0];
  const unsigned kld1 = (unsigned)(uintptr_t)&Ksh[krow1 * 64 + kcol1];
  const unsigned vld0 = (unsigned)(uintptr_t)&Vsh[vrow0 * 32 + vcol0];
  const unsigned vld1 = (unsigned)(uintptr_t)&Vsh[vrow1 * 32 + vcol1];
  const unsigned kmo0 = (unsigned)(krow0 * Dc + kcol0) * 2u;  // byte offsets
  const unsigned kmo1 = (unsigned)(krow1 * Dc + kcol1) * 2u;
  const unsigned vmo0 = (unsigned)(vrow0 * Sc + vcol0) * 2u;
  const unsigned vmo1 = (unsigned)(vrow1 * Sc + vcol1) * 2u;

  for (int kt = 0; kt < Sc / 32; ++kt) {       // 64 iterations of 32 keys
    const int key0 = kt * 32;

    // --- cooperative async stage of K/V tiles into LDS (ASYNCcnt path) ---
    {
      const unsigned long long kbase = (unsigned long long)(uintptr_t)(
          Kb + (size_t)(b * Sc + key0) * Dc + h * HDc);
      const unsigned long long vbase = (unsigned long long)(uintptr_t)(
          Vt + vtbase + key0);
      asm volatile("global_load_async_to_lds_b128 %0, %1, %2"
                   :: "v"(kld0), "v"(kmo0), "s"(kbase) : "memory");
      asm volatile("global_load_async_to_lds_b128 %0, %1, %2"
                   :: "v"(kld1), "v"(kmo1), "s"(kbase) : "memory");
      asm volatile("global_load_async_to_lds_b128 %0, %1, %2"
                   :: "v"(vld0), "v"(vmo0), "s"(vbase) : "memory");
      asm volatile("global_load_async_to_lds_b128 %0, %1, %2"
                   :: "v"(vld1), "v"(vmo1), "s"(vbase) : "memory");
      asm volatile("s_wait_asynccnt 0" ::: "memory");
    }
    __syncthreads();

    // --- scores: S = Q * K^T (B-frags from LDS: contiguous hd runs) ---
    FragC s0, s1;
#pragma unroll
    for (int j = 0; j < 8; ++j) { s0.f[j] = 0.f; s1.f[j] = 0.f; }
    {
      const unsigned short* kp0 = &Ksh[(size_t)m_id * 64];         // tile 0
      const unsigned short* kp1 = &Ksh[(size_t)(16 + m_id) * 64];  // tile 1
      FragAB kb0, kb1, kb2, kb3;
      load_frag16(kb0, kp0 + g * 16, kp0 + g * 16 + 8);        // hd 0..31
      load_frag16(kb1, kp0 + 32 + g * 16, kp0 + 32 + g * 16 + 8); // hd 32..63
      load_frag16(kb2, kp1 + g * 16, kp1 + g * 16 + 8);
      load_frag16(kb3, kp1 + 32 + g * 16, kp1 + 32 + g * 16 + 8);
      s0.v = __builtin_amdgcn_wmma_f32_16x16x32_bf16(
          false, qa[0].v, false, kb0.v, (short)0, s0.v, false, false);
      s0.v = __builtin_amdgcn_wmma_f32_16x16x32_bf16(
          false, qa[1].v, false, kb1.v, (short)0, s0.v, false, false);
      s1.v = __builtin_amdgcn_wmma_f32_16x16x32_bf16(
          false, qa[0].v, false, kb2.v, (short)0, s1.v, false, false);
      s1.v = __builtin_amdgcn_wmma_f32_16x16x32_bf16(
          false, qa[1].v, false, kb3.v, (short)0, s1.v, false, false);
    }

    // --- online softmax across the 32 columns ---
    float p0[8], p1[8], rsc[8];
#pragma unroll
    for (int j = 0; j < 8; ++j) {
      float v0 = s0.f[j] * 0.125f;   // 1/sqrt(64)
      float v1 = s1.f[j] * 0.125f;
      float t = fmaxf(v0, v1);
      t = fmaxf(t, __shfl_xor(t, 1, 32));
      t = fmaxf(t, __shfl_xor(t, 2, 32));
      t = fmaxf(t, __shfl_xor(t, 4, 32));
      t = fmaxf(t, __shfl_xor(t, 8, 32));
      float mn = fmaxf(mrun[j], t);
      rsc[j] = __expf(mrun[j] - mn);
      mrun[j] = mn;
      p0[j] = __expf(v0 - mn);
      p1[j] = __expf(v1 - mn);
      float rs = p0[j] + p1[j];
      rs += __shfl_xor(rs, 1, 32);
      rs += __shfl_xor(rs, 2, 32);
      rs += __shfl_xor(rs, 4, 32);
      rs += __shfl_xor(rs, 8, 32);
      lrun[j] = lrun[j] * rsc[j] + rs;
    }
#pragma unroll
    for (int c = 0; c < 4; ++c)
#pragma unroll
      for (int j = 0; j < 8; ++j) o[c].f[j] *= rsc[j];

    // --- C-frag -> A-frag layout conversion through LDS (same wave only) ---
#pragma unroll
    for (int j = 0; j < 8; ++j) {
      const int r = j + 8 * g;
      pl[r * 40 + m_id]      = f2bf(p0[j]);
      pl[r * 40 + 16 + m_id] = f2bf(p1[j]);
    }
    asm volatile("s_wait_dscnt 0" ::: "memory");
    FragAB pa;
    {
      const unsigned short* pr = pl + m_id * 40 + g * 16;
      pa.q[0] = *reinterpret_cast<const uint4*>(pr);
      pa.q[1] = *reinterpret_cast<const uint4*>(pr + 8);
    }

    // --- O += P * V (V B-frags from LDS: contiguous key runs) ---
#pragma unroll
    for (int c = 0; c < 4; ++c) {
      const unsigned short* vp = &Vsh[(size_t)(c * 16 + m_id) * 32 + g * 16];
      FragAB vb;
      load_frag16(vb, vp, vp + 8);
      o[c].v = __builtin_amdgcn_wmma_f32_16x16x32_bf16(
          false, pa.v, false, vb.v, (short)0, o[c].v, false, false);
    }
    __syncthreads();   // WAR: next iteration overwrites Ksh/Vsh
  }

  // normalize + store ctx (bf16, token-major)
#pragma unroll
  for (int j = 0; j < 8; ++j) {
    const float inv = 1.f / lrun[j];
    const size_t row = (size_t)(b * Sc + qt * 16 + j + 8 * g);
#pragma unroll
    for (int c = 0; c < 4; ++c) {
      const int col = h * HDc + c * 16 + m_id;
      ctx[row * Dc + col] = f2bf(o[c].f[j] * inv);
    }
  }
}

// ---------------- host-side orchestration ----------------
extern "C" void kernel_launch(void* const* d_in, const int* in_sizes, int n_in,
                              void* d_out, int out_size, void* d_ws, size_t ws_size,
                              hipStream_t stream) {
  (void)in_sizes; (void)n_in; (void)out_size; (void)ws_size;
  const float* q   = (const float*)d_in[0];
  const float* k   = (const float*)d_in[1];
  const float* v   = (const float*)d_in[2];
  const float* Wq  = (const float*)d_in[3];
  const float* bq  = (const float*)d_in[4];
  const float* Wk  = (const float*)d_in[5];
  const float* bk  = (const float*)d_in[6];
  const float* Wv  = (const float*)d_in[7];
  const float* bv  = (const float*)d_in[8];
  const float* Wo  = (const float*)d_in[9];
  const float* bo  = (const float*)d_in[10];
  const float* Wad = (const float*)d_in[11];
  const float* bad = (const float*)d_in[12];
  const float* Wau = (const float*)d_in[13];
  const float* bau = (const float*)d_in[14];
  float* out = (float*)d_out;

  // workspace carve-up (bf16 staging + fp32 intermediates), ~94 MB (L2-resident)
  unsigned short* p = reinterpret_cast<unsigned short*>(d_ws);
  unsigned short* XQ  = p; p += NTD;
  unsigned short* XK  = p; p += NTD;
  unsigned short* XV  = p; p += NTD;
  unsigned short* WQb = p; p += (size_t)Dc * Dc;
  unsigned short* WKb = p; p += (size_t)Dc * Dc;
  unsigned short* WVb = p; p += (size_t)Dc * Dc;
  unsigned short* WOb = p; p += (size_t)Dc * Dc;
  unsigned short* WADb = p; p += (size_t)ADc * Dc;
  unsigned short* WAUb = p; p += (size_t)Dc * ADc;
  unsigned short* QH  = p; p += NTD;
  unsigned short* KH  = p; p += NTD;
  unsigned short* VT  = p; p += NTD;
  unsigned short* CTX = p; p += NTD;
  unsigned short* OUTb = p; p += NTD;
  unsigned short* Gb  = p; p += (size_t)NTc * ADc;
  float* OUTf = reinterpret_cast<float*>(p);

  // fp32 -> bf16 staging
  const dim3 cb(256), cg(2048);
  cvt_bf16_kernel<<<cg, cb, 0, stream>>>(q,  XQ,  (int)NTD);
  cvt_bf16_kernel<<<cg, cb, 0, stream>>>(k,  XK,  (int)NTD);
  cvt_bf16_kernel<<<cg, cb, 0, stream>>>(v,  XV,  (int)NTD);
  cvt_bf16_kernel<<<cg, cb, 0, stream>>>(Wq, WQb, Dc * Dc);
  cvt_bf16_kernel<<<cg, cb, 0, stream>>>(Wk, WKb, Dc * Dc);
  cvt_bf16_kernel<<<cg, cb, 0, stream>>>(Wv, WVb, Dc * Dc);
  cvt_bf16_kernel<<<cg, cb, 0, stream>>>(Wo, WOb, Dc * Dc);
  cvt_bf16_kernel<<<cg, cb, 0, stream>>>(Wad, WADb, ADc * Dc);
  cvt_bf16_kernel<<<cg, cb, 0, stream>>>(Wau, WAUb, Dc * ADc);

  const dim3 blk(128);
  const dim3 gD(Dc / 64, NTc / 128);    // (16, 32) for N=1024
  const dim3 gA(ADc / 64, NTc / 128);   // (2, 32) for N=128

  // Q/K/V projections
  gemm_bf16_kernel<<<gD, blk, 0, stream>>>(XQ, Dc, WQb, Dc, bq,
      NTc, Dc, Dc, MODE_BF16, QH, nullptr, nullptr);
  gemm_bf16_kernel<<<gD, blk, 0, stream>>>(XK, Dc, WKb, Dc, bk,
      NTc, Dc, Dc, MODE_BF16, KH, nullptr, nullptr);
  gemm_bf16_kernel<<<gD, blk, 0, stream>>>(XV, Dc, WVb, Dc, bv,
      NTc, Dc, Dc, MODE_VT, VT, nullptr, nullptr);

  // attention: 1024 blocks x 4 waves (block-uniform (b,h))
  attn_kernel<<<dim3(1024), blk, 0, stream>>>(QH, KH, VT, CTX);

  // output projection (fp32 kept for residual, bf16 copy feeds adapter)
  gemm_bf16_kernel<<<gD, blk, 0, stream>>>(CTX, Dc, WOb, Dc, bo,
      NTc, Dc, Dc, MODE_F32BF16, OUTb, OUTf, nullptr);

  // adapter down + exact GELU
  gemm_bf16_kernel<<<gA, blk, 0, stream>>>(OUTb, Dc, WADb, Dc, bad,
      NTc, ADc, Dc, MODE_GELU, Gb, nullptr, nullptr);

  // adapter up + residual -> final fp32 output
  gemm_bf16_kernel<<<gD, blk, 0, stream>>>(Gb, ADc, WAUb, ADc, bau,
      NTc, Dc, ADc, MODE_RESID, nullptr, out, OUTf);
}
